// MoELayer_41120016892663
// MI455X (gfx1250) — matrix operand order repaired and checked by reading, not compiled, not run on hardware
//
#include <hip/hip_runtime.h>

// ---------------- problem constants ----------------
constexpr int NT   = 2048;   // tokens
constexpr int DM   = 1024;   // model dim
constexpr int NH   = 16;     // heads
constexpr int DH   = 64;     // head dim
constexpr int FF   = 4096;   // expert hidden
constexpr int NE   = 16;     // experts
constexpr int CAPE = 160;    // int(1.25*2048/16)

typedef __attribute__((ext_vector_type(16))) _Float16 v16h;
typedef __attribute__((ext_vector_type(8)))  float    v8f;
typedef unsigned int u32x4 __attribute__((ext_vector_type(4)));
typedef int          i32x4 __attribute__((ext_vector_type(4)));
typedef int          i32x8 __attribute__((ext_vector_type(8)));

union H8   { _Float16 h[8]; uint4 q; };
union F16U { v16h v; uint4 q[2]; };

// A-frag (16x32 f16): lane L holds row M=L%16; halves {sel..sel+7, sel+16..sel+23}, sel=(L<16)?0:8
__device__ inline v16h ldFragA(const _Float16* rowp, int sel) {
  F16U r;
  r.q[0] = *(const uint4*)(rowp + sel);
  r.q[1] = *(const uint4*)(rowp + sel + 16);
  return r.v;
}
// B-frag (32x16 f16) from [n][k] storage: lane L holds col N=L%16; halves {kbase..kbase+7, kbase+16..kbase+23}, kbase=(L<16)?0:8
__device__ inline v16h ldFragB(const _Float16* rowp, int kbase) {
  F16U r;
  r.q[0] = *(const uint4*)(rowp + kbase);
  r.q[1] = *(const uint4*)(rowp + kbase + 16);
  return r.v;
}

// ---------------- TDM descriptor issue (A tile: 128 rows x 32 halves, padded rows of 40) ----
#if __has_builtin(__builtin_amdgcn_tensor_load_to_lds)
#define HAVE_TDM 1
__device__ inline void tdm_load_a(const _Float16* gsrc, unsigned ldsByteOff,
                                  int kRemain, int rowsAvail, int K) {
  const unsigned long long ga = (unsigned long long)(uintptr_t)gsrc;
  u32x4 g0; i32x8 g1; i32x4 g2; i32x4 g3;
  g0[0] = 1u;                                                  // count=1, user-mode D#
  g0[1] = ldsByteOff;                                          // lds_addr (bytes)
  g0[2] = (unsigned)(ga & 0xFFFFFFFFu);                        // global_addr[31:0]
  g0[3] = (unsigned)((ga >> 32) & 0x01FFFFFFu) | (2u << 30);   // global_addr[56:32] | type=2
  // data_size=2B(code1), pad_enable, pad_interval=16 DWORDs(code3), pad_amount=4 DWORDs(code3)
  g1[0] = (int)((1u << 16) | (1u << 20) | (3u << 22) | (3u << 25));
  g1[1] = (int)(((unsigned)kRemain & 0xFFFFu) << 16);                                   // tensor_dim0 lo16
  g1[2] = (int)((((unsigned)kRemain >> 16) & 0xFFFFu) | (((unsigned)rowsAvail & 0xFFFFu) << 16)); // dim0 hi | dim1 lo
  g1[3] = (int)((((unsigned)rowsAvail >> 16) & 0xFFFFu) | (32u << 16));                 // dim1 hi | tile_dim0=32
  g1[4] = 128;                                                 // tile_dim1=128 (tile_dim2=0)
  g1[5] = K;                                                   // tensor_dim0_stride (elements)
  g1[6] = 0;
  g1[7] = 0;
  g2[0] = g2[1] = g2[2] = g2[3] = 0;
  g3[0] = g3[1] = g3[2] = g3[3] = 0;
#if defined(__has_include)
#if __has_include(<hip/amd_detail/amd_gfx1250_TDM.h>)
  i32x8 g4; g4[0]=g4[1]=g4[2]=g4[3]=g4[4]=g4[5]=g4[6]=g4[7]=0;
  __builtin_amdgcn_tensor_load_to_lds(g0, g1, g2, g3, g4, 0);  // 6-arg toolchain
#else
  __builtin_amdgcn_tensor_load_to_lds(g0, g1, g2, g3, 0);      // 5-arg toolchain (ROCm 7.2)
#endif
#else
  __builtin_amdgcn_tensor_load_to_lds(g0, g1, g2, g3, 0);
#endif
}
#else
#define HAVE_TDM 0
#endif

// ---------------- generic WMMA GEMM ----------------
// C[m][n] = sum_k A[m][k] * W[n][k]  (A @ W.T), optional bias/relu/residual epilogue.
// A: f16 row-major [M x K] (TDM-staged into LDS); W: f32 row-major [Nout x K]
// (converted fp32->f16 in VGPR pass-through -- TDM cannot convert, and on-the-fly
// conversion is bandwidth-optimal for once-streamed MoE weights).
// Block tile 128x128, K-step 32, double-buffered LDS. 8 waves (2x4), wave tile 64x32.
constexpr int LDS_TILE_H = 128 * 40;                       // halves per (padded) tile buffer
constexpr unsigned GEMM_LDS_BYTES = 4u * LDS_TILE_H * 2u;  // A0,A1,B0,B1

template<bool BIAS, bool RELU, bool ADDSRC, bool OUTF32, bool OUTF16>
__global__ __launch_bounds__(256)
void wmma_gemm(const _Float16* __restrict__ A, long sA,
               const float* __restrict__ W, long sW,
               const float* __restrict__ bias, long sBias,
               const float* __restrict__ addsrc,
               float* __restrict__ Cf, _Float16* __restrict__ Ch, long sC,
               int M, int Nout, int K)
{
  extern __shared__ __align__(16) _Float16 smem[];   // [A0][A1][B0][B1], rows padded to 40 halves

  const int bz = blockIdx.z;
  A += (long)bz * sA;  W += (long)bz * sW;
  if (BIAS)   bias += (long)bz * sBias;
  if (OUTF32) Cf   += (long)bz * sC;
  if (OUTF16) Ch   += (long)bz * sC;

  const int tid = threadIdx.x;
  const int m0g = blockIdx.x * 128;
  const int n0g = blockIdx.y * 128;

  const int wv = tid >> 5, ln = tid & 31;
  const int m0 = (wv >> 2) * 64, n0 = (wv & 3) * 32;
  const int lh = ln & 15;
  const bool hi = ln >= 16;
  const int rowsAvail = M - m0g;

  v8f acc[4][2] = {};

  // staging indices: 128 rows x 2 segments (A: 16 halves, B: 16 floats) per thread
  const int srow = tid >> 1, sseg = tid & 1;
  const int agr = m0g + srow;
  const int bgr = n0g + srow;

  const int nk = K >> 5;
  float breg[16];

  // buffer addressing via integer offsets (NO pointer arrays of LDS addresses:
  // those get folded into addrspacecast static initializers that ld.lld rejects)
  auto bufAp = [&](int buf) -> _Float16* { return smem + buf * LDS_TILE_H; };
  auto bufBp = [&](int buf) -> _Float16* { return smem + (2 + buf) * LDS_TILE_H; };

  auto stageB_load = [&](int kt) {
    const float* src = W + (long)bgr * K + (kt << 5) + sseg * 16;
    const bool ok = bgr < Nout;
    #pragma unroll
    for (int i = 0; i < 16; i++) breg[i] = ok ? src[i] : 0.f;
  };
  auto stageB_store = [&](int buf) {
    H8 h0, h1;
    #pragma unroll
    for (int i = 0; i < 8; i++) h0.h[i] = (_Float16)breg[i];
    #pragma unroll
    for (int i = 0; i < 8; i++) h1.h[i] = (_Float16)breg[8 + i];
    _Float16* dst = bufBp(buf) + srow * 40 + sseg * 16;
    *(uint4*)(dst)     = h0.q;
    *(uint4*)(dst + 8) = h1.q;
  };
#if !HAVE_TDM
  auto stageA_sync = [&](int kt, int buf) {
    uint4 v0 = make_uint4(0u,0u,0u,0u), v1 = v0;
    if (agr < M) {
      const _Float16* src = A + (long)agr * K + (kt << 5) + sseg * 16;
      v0 = *(const uint4*)(src);
      v1 = *(const uint4*)(src + 8);
    }
    _Float16* dst = bufAp(buf) + srow * 40 + sseg * 16;
    *(uint4*)(dst)     = v0;
    *(uint4*)(dst + 8) = v1;
  };
#endif

  // ---- prologue: stage tile 0 ----
  stageB_load(0);
#if HAVE_TDM
  if (wv == 0)
    tdm_load_a(A + (long)m0g * K, 0u, K, rowsAvail, K);
#else
  stageA_sync(0, 0);
#endif
  stageB_store(0);
#if HAVE_TDM
  if (wv == 0) __builtin_amdgcn_s_wait_tensorcnt(0);
#endif
  __syncthreads();

  for (int kt = 0; kt < nk; ++kt) {
    const int cur = kt & 1, nxt = cur ^ 1;
    const bool more = (kt + 1) < nk;

    if (more) {
      stageB_load(kt + 1);                         // global f32 loads in flight during WMMA
#if HAVE_TDM
      if (wv == 0)
        tdm_load_a(A + (long)m0g * K + ((kt + 1) << 5),
                   (unsigned)nxt * (unsigned)LDS_TILE_H * 2u,
                   K - ((kt + 1) << 5), rowsAvail, K);
#endif
    }

    const int aSel = hi ? 8 : 0;
    const int bSel = hi ? 8 : 0;
    v16h aF[4], bF[2];
    #pragma unroll
    for (int i = 0; i < 4; i++) aF[i] = ldFragA(bufAp(cur) + (m0 + i*16 + lh) * 40, aSel);
    #pragma unroll
    for (int j = 0; j < 2; j++) bF[j] = ldFragB(bufBp(cur) + (n0 + j*16 + lh) * 40, bSel);
    #pragma unroll
    for (int i = 0; i < 4; i++)
      #pragma unroll
      for (int j = 0; j < 2; j++)
        acc[i][j] = __builtin_amdgcn_wmma_f32_16x16x32_f16(
            false, aF[i], false, bF[j], (short)0, acc[i][j], false, false);

    if (more) {
      stageB_store(nxt);
#if !HAVE_TDM
      stageA_sync(kt + 1, nxt);
#else
      if (wv == 0) __builtin_amdgcn_s_wait_tensorcnt(0);
#endif
    }
    __syncthreads();
  }

  // ---- epilogue ----
  #pragma unroll
  for (int i = 0; i < 4; i++) {
    #pragma unroll
    for (int j = 0; j < 2; j++) {
      const int col = n0g + n0 + j * 16 + lh;
      float bval = 0.f;
      if (BIAS) bval = bias[col];
      #pragma unroll
      for (int v = 0; v < 8; v++) {
        const int row = m0g + m0 + i * 16 + v + (hi ? 8 : 0);
        if (row < M) {
          float val = acc[i][j][v] + bval;
          if (RELU) val = fmaxf(val, 0.f);
          const long idx = (long)row * Nout + col;
          if (ADDSRC) val += addsrc[idx];
          if (OUTF32) Cf[idx] = val;
          if (OUTF16) Ch[idx] = (_Float16)val;
        }
      }
    }
  }
}

// ---------------- LayerNorm (one block per row) ----------------
__global__ __launch_bounds__(256)
void ln_kernel(const float* __restrict__ x, const float* __restrict__ w,
               const float* __restrict__ b, float* __restrict__ of,
               _Float16* __restrict__ oh)
{
  __shared__ float red[256];
  const int row = blockIdx.x;
  const float* xr = x + (long)row * DM;
  float vals[4];
  float s = 0.f;
  #pragma unroll
  for (int i = 0; i < 4; i++) { vals[i] = xr[threadIdx.x + 256 * i]; s += vals[i]; }
  red[threadIdx.x] = s; __syncthreads();
  for (int st = 128; st > 0; st >>= 1) { if (threadIdx.x < st) red[threadIdx.x] += red[threadIdx.x + st]; __syncthreads(); }
  const float mu = red[0] * (1.f / DM);
  __syncthreads();
  float s2 = 0.f;
  #pragma unroll
  for (int i = 0; i < 4; i++) { float d = vals[i] - mu; s2 += d * d; }
  red[threadIdx.x] = s2; __syncthreads();
  for (int st = 128; st > 0; st >>= 1) { if (threadIdx.x < st) red[threadIdx.x] += red[threadIdx.x + st]; __syncthreads(); }
  const float rinv = rsqrtf(red[0] * (1.f / DM) + 1e-5f);
  #pragma unroll
  for (int i = 0; i < 4; i++) {
    const int d = threadIdx.x + 256 * i;
    const float y = (vals[i] - mu) * rinv * w[d] + b[d];
    if (of) of[(long)row * DM + d] = y;
    if (oh) oh[(long)row * DM + d] = (_Float16)y;
  }
}

// ---------------- V transpose: VT[h*64+dh][n] = V[n][h*64+dh] ----------------
__global__ __launch_bounds__(256)
void vtrans_kernel(const _Float16* __restrict__ V, _Float16* __restrict__ VT)
{
  const int r = blockIdx.x; // h*64 + dh
  #pragma unroll
  for (int i = 0; i < 8; i++) {
    const int n = threadIdx.x + 256 * i;
    VT[(long)r * NT + n] = V[(long)n * DM + r];
  }
}

// ---------------- Flash attention: 1 wave per (head, 16 query rows) ----------------
__global__ __launch_bounds__(256)
void flash_attn(const _Float16* __restrict__ Q, const _Float16* __restrict__ K,
                const _Float16* __restrict__ VT, _Float16* __restrict__ O)
{
  __shared__ __align__(16) _Float16 pBuf[8][16][40];
  const int wv = threadIdx.x >> 5, ln = threadIdx.x & 31;
  const int job = blockIdx.x * 8 + wv;      // 2048 jobs = 16 heads * 128 qblocks
  const int h  = job >> 7;
  const int q0 = (job & 127) << 4;
  const int lh = ln & 15;
  const bool hi = ln >= 16;

  v16h aQ[2];
  {
    const _Float16* qr = Q + (long)(q0 + lh) * DM + h * DH;
    #pragma unroll
    for (int ki = 0; ki < 2; ki++) {
      const int base = ki * 32 + (hi ? 8 : 0);
      F16U r;
      r.q[0] = *(const uint4*)(qr + base);
      r.q[1] = *(const uint4*)(qr + base + 16);
      aQ[ki] = r.v;
    }
  }

  v8f oAcc[4] = {};
  float mPrev[8], lSum[8];
  #pragma unroll
  for (int v = 0; v < 8; v++) { mPrev[v] = -3.0e38f; lSum[v] = 0.f; }

  const int kend = q0 + 16;
  for (int kb = 0; kb < kend; kb += 32) {
    float s[2][8];
    #pragma unroll
    for (int j = 0; j < 2; j++) {
      v8f sAcc = {};
      const _Float16* kr = K + (long)(kb + j * 16 + lh) * DM + h * DH;
      #pragma unroll
      for (int ki = 0; ki < 2; ki++) {
        const int base = ki * 32 + (hi ? 8 : 0);
        F16U r;
        r.q[0] = *(const uint4*)(kr + base);
        r.q[1] = *(const uint4*)(kr + base + 16);
        sAcc = __builtin_amdgcn_wmma_f32_16x16x32_f16(
            false, aQ[ki], false, r.v, (short)0, sAcc, false, false);
      }
      const int key = kb + j * 16 + lh;
      #pragma unroll
      for (int v = 0; v < 8; v++) {
        const int qrow = q0 + v + (hi ? 8 : 0);
        s[j][v] = (key > qrow) ? -3.0e38f : sAcc[v] * 0.125f;   // 1/sqrt(64)
      }
    }
    float alpha[8];
    #pragma unroll
    for (int v = 0; v < 8; v++) {
      float m = fmaxf(s[0][v], s[1][v]);
      for (int off = 8; off > 0; off >>= 1) m = fmaxf(m, __shfl_xor(m, off, 32));
      const float mNew = fmaxf(mPrev[v], m);
      alpha[v] = expf(mPrev[v] - mNew);
      const float p0 = expf(s[0][v] - mNew);
      const float p1 = expf(s[1][v] - mNew);
      float ps = p0 + p1;
      for (int off = 8; off > 0; off >>= 1) ps += __shfl_xor(ps, off, 32);
      lSum[v] = lSum[v] * alpha[v] + ps;
      mPrev[v] = mNew;
      const int ql = v + (hi ? 8 : 0);
      pBuf[wv][ql][lh]      = (_Float16)p0;
      pBuf[wv][ql][16 + lh] = (_Float16)p1;
    }
    #pragma unroll
    for (int f = 0; f < 4; f++)
      #pragma unroll
      for (int v = 0; v < 8; v++) oAcc[f][v] *= alpha[v];

    v16h aP;
    {
      const _Float16* pr = &pBuf[wv][lh][0];
      const int base = hi ? 8 : 0;
      F16U r;
      r.q[0] = *(const uint4*)(pr + base);
      r.q[1] = *(const uint4*)(pr + base + 16);
      aP = r.v;
    }
    #pragma unroll
    for (int f = 0; f < 4; f++) {
      const _Float16* vr = VT + ((long)h * DH + f * 16 + lh) * NT + kb + (hi ? 8 : 0);
      F16U r;
      r.q[0] = *(const uint4*)(vr);
      r.q[1] = *(const uint4*)(vr + 16);
      oAcc[f] = __builtin_amdgcn_wmma_f32_16x16x32_f16(
          false, aP, false, r.v, (short)0, oAcc[f], false, false);
    }
  }

  #pragma unroll
  for (int f = 0; f < 4; f++) {
    const int dh = f * 16 + lh;
    #pragma unroll
    for (int v = 0; v < 8; v++) {
      const int qrow = q0 + v + (hi ? 8 : 0);
      O[(long)qrow * DM + h * DH + dh] = (_Float16)(oAcc[f][v] / lSum[v]);
    }
  }
}

// ---------------- router: one wave per token ----------------
__global__ __launch_bounds__(256)
void router_kernel(const float* __restrict__ h2, const float* __restrict__ rw,
                   const float* __restrict__ rb, int* __restrict__ routes,
                   float* __restrict__ pmax)
{
  const int wv = threadIdx.x >> 5, ln = threadIdx.x & 31;
  const int t = blockIdx.x * 8 + wv;
  const int e = ln & 15, half = ln >> 4;
  const float* hr = h2 + (long)t * DM;
  const float* wr = rw + (long)e * DM;
  float acc = 0.f;
  const int i0 = half * 512;
  for (int i = i0; i < i0 + 512; ++i) acc += hr[i] * wr[i];
  acc += __shfl_xor(acc, 16, 32);
  const float logit = acc + rb[e];
  float mv = logit; int mi = e;
  for (int off = 8; off > 0; off >>= 1) {
    const float ov = __shfl_xor(mv, off, 32);
    const int   oi = __shfl_xor(mi, off, 32);
    if (ov > mv || (ov == mv && oi < mi)) { mv = ov; mi = oi; }
  }
  float z = expf(logit - mv);
  for (int off = 8; off > 0; off >>= 1) z += __shfl_xor(z, off, 32);
  if (ln == 0) { routes[t] = mi; pmax[t] = 1.f / z; }
}

// ---------------- deterministic capacity scan (1 wave) ----------------
__global__ void route_scan(const int* __restrict__ routes, int* __restrict__ keep,
                           int* __restrict__ slot, int* __restrict__ inv)
{
  const int tid = threadIdx.x;
  for (int i = tid; i < NE * CAPE; i += 32) inv[i] = -1;
  __syncthreads();
  if (tid < NE) {
    int cnt = 0;
    for (int t = 0; t < NT; ++t) {
      if (routes[t] == tid) {
        if (cnt < CAPE) { const int s = tid * CAPE + cnt; slot[t] = s; keep[t] = 1; inv[s] = t; }
        else            { slot[t] = -1; keep[t] = 0; }
        cnt++;
      }
    }
  }
}

// ---------------- gather tokens into expert slots ----------------
__global__ __launch_bounds__(256)
void gather_kernel(const _Float16* __restrict__ h2h, const int* __restrict__ inv,
                   _Float16* __restrict__ xe)
{
  const int r = blockIdx.x;
  const int t = inv[r];
  #pragma unroll
  for (int i = 0; i < 4; i++) {
    const int d = threadIdx.x + 256 * i;
    xe[(long)r * DM + d] = (t >= 0) ? h2h[(long)t * DM + d] : (_Float16)0.f;
  }
}

// ---------------- final combine ----------------
__global__ __launch_bounds__(256)
void final_kernel(const float* __restrict__ x2, const float* __restrict__ h2f,
                  const float* __restrict__ ye, const int* __restrict__ keep,
                  const int* __restrict__ slot, const float* __restrict__ pmax,
                  float* __restrict__ out)
{
  const int t = blockIdx.x;
  const int k = keep[t], s = slot[t];
  const float pm = pmax[t];
  #pragma unroll
  for (int i = 0; i < 4; i++) {
    const int d = threadIdx.x + 256 * i;
    const float y = k ? ye[(long)s * DM + d] : h2f[(long)t * DM + d];
    out[(long)t * DM + d] = x2[(long)t * DM + d] + pm * y;
  }
}

// ---------------- host launcher ----------------
extern "C" void kernel_launch(void* const* d_in, const int* in_sizes, int n_in,
                              void* d_out, int out_size, void* d_ws, size_t ws_size,
                              hipStream_t stream)
{
  (void)in_sizes; (void)n_in; (void)out_size;
  const float* x    = (const float*)d_in[0];
  const float* wk   = (const float*)d_in[1];
  const float* bk   = (const float*)d_in[2];
  const float* wq   = (const float*)d_in[3];
  const float* bq   = (const float*)d_in[4];
  const float* wv_  = (const float*)d_in[5];
  const float* bv   = (const float*)d_in[6];
  const float* ipw  = (const float*)d_in[7];
  const float* ipb  = (const float*)d_in[8];
  const float* opw  = (const float*)d_in[9];
  const float* opb  = (const float*)d_in[10];
  const float* ln1w = (const float*)d_in[11];
  const float* ln1b = (const float*)d_in[12];
  const float* ln2w = (const float*)d_in[13];
  const float* ln2b = (const float*)d_in[14];
  const float* rw   = (const float*)d_in[15];
  const float* rb   = (const float*)d_in[16];
  const float* w1   = (const float*)d_in[17];
  const float* b1   = (const float*)d_in[18];
  const float* w2   = (const float*)d_in[19];
  const float* b2   = (const float*)d_in[20];
  float* out = (float*)d_out;

  char* ws = (char*)d_ws;
  size_t off = 0;
  auto carve = [&](size_t bytes) -> char* {
    off = (off + 255) & ~(size_t)255;
    char* p = ws + off;
    off += bytes;
    return p;
  };

  _Float16* hln1 = (_Float16*)carve((size_t)NT * DM * 2);
  _Float16* qin  = (_Float16*)carve((size_t)NT * DM * 2);
  _Float16* kin  = (_Float16*)carve((size_t)NT * DM * 2);
  _Float16* vin  = (_Float16*)carve((size_t)NT * DM * 2);
  _Float16* qf   = (_Float16*)carve((size_t)NT * DM * 2);
  _Float16* kf   = (_Float16*)carve((size_t)NT * DM * 2);
  _Float16* vf   = (_Float16*)carve((size_t)NT * DM * 2);
  _Float16* vt   = (_Float16*)carve((size_t)NT * DM * 2);
  _Float16* ctx  = (_Float16*)carve((size_t)NT * DM * 2);
  float*    x2   = (float*)   carve((size_t)NT * DM * 4);
  float*    h2f  = (float*)   carve((size_t)NT * DM * 4);
  _Float16* h2h  = (_Float16*)carve((size_t)NT * DM * 2);
  int*      routes = (int*)   carve((size_t)NT * 4);
  float*    pmaxA  = (float*) carve((size_t)NT * 4);
  int*      keep   = (int*)   carve((size_t)NT * 4);
  int*      slotA  = (int*)   carve((size_t)NT * 4);
  int*      inv    = (int*)   carve((size_t)NE * CAPE * 4);
  _Float16* xe   = (_Float16*)carve((size_t)NE * CAPE * DM * 2);
  _Float16* h1m  = (_Float16*)carve((size_t)NE * CAPE * FF * 2);
  float*    ye   = (float*)   carve((size_t)NE * CAPE * DM * 4);
  if (off > ws_size) return; // insufficient workspace

  const dim3 blk(256);
  const dim3 gProj(NT / 128, DM / 128, 1);
  const unsigned shmem = GEMM_LDS_BYTES;

  // 1) h = LN1(x) (f16)
  ln_kernel<<<NT, blk, 0, stream>>>(x, ln1w, ln1b, nullptr, hln1);

  // 2) first projections (note reference swap: q from wk, k from wq)
  wmma_gemm<true,false,false,false,true><<<gProj, blk, shmem, stream>>>(
      hln1, 0, wk, 0, bk, 0, nullptr, nullptr, qin, 0, NT, DM, DM);
  wmma_gemm<true,false,false,false,true><<<gProj, blk, shmem, stream>>>(
      hln1, 0, wq, 0, bq, 0, nullptr, nullptr, kin, 0, NT, DM, DM);
  wmma_gemm<true,false,false,false,true><<<gProj, blk, shmem, stream>>>(
      hln1, 0, wv_, 0, bv, 0, nullptr, nullptr, vin, 0, NT, DM, DM);

  // 3) second projections through ipw slices
  wmma_gemm<true,false,false,false,true><<<gProj, blk, shmem, stream>>>(
      qin, 0, ipw, 0, ipb, 0, nullptr, nullptr, qf, 0, NT, DM, DM);
  wmma_gemm<true,false,false,false,true><<<gProj, blk, shmem, stream>>>(
      kin, 0, ipw + (size_t)DM * DM, 0, ipb + DM, 0, nullptr, nullptr, kf, 0, NT, DM, DM);
  wmma_gemm<true,false,false,false,true><<<gProj, blk, shmem, stream>>>(
      vin, 0, ipw + (size_t)2 * DM * DM, 0, ipb + 2 * DM, 0, nullptr, nullptr, vf, 0, NT, DM, DM);

  // 4) flash attention
  vtrans_kernel<<<NH * DH, blk, 0, stream>>>(vf, vt);
  flash_attn<<<(NH * (NT / 16)) / 8, blk, 0, stream>>>(qf, kf, vt, ctx);

  // 5) out projection + residual -> x2 (f32)
  wmma_gemm<true,false,true,true,false><<<gProj, blk, shmem, stream>>>(
      ctx, 0, opw, 0, opb, 0, x, x2, nullptr, 0, NT, DM, DM);

  // 6) LN2 -> h2 (f32 + f16)
  ln_kernel<<<NT, blk, 0, stream>>>(x2, ln2w, ln2b, h2f, h2h);

  // 7) routing
  router_kernel<<<NT / 8, blk, 0, stream>>>(h2f, rw, rb, routes, pmaxA);
  route_scan<<<1, 32, 0, stream>>>(routes, keep, slotA, inv);
  gather_kernel<<<NE * CAPE, blk, 0, stream>>>(h2h, inv, xe);

  // 8) expert FFN (batched over experts)
  wmma_gemm<true,true,false,false,true><<<dim3((CAPE + 127) / 128, FF / 128, NE), blk, shmem, stream>>>(
      xe, (long)CAPE * DM, w1, (long)FF * DM, b1, FF, nullptr,
      nullptr, h1m, (long)CAPE * FF, CAPE, FF, DM);
  wmma_gemm<true,false,false,true,false><<<dim3((CAPE + 127) / 128, DM / 128, NE), blk, shmem, stream>>>(
      h1m, (long)CAPE * FF, w2, (long)DM * FF, b2, DM, nullptr,
      ye, nullptr, (long)CAPE * DM, CAPE, DM, FF);

  // 9) combine
  final_kernel<<<NT, blk, 0, stream>>>(x2, h2f, ye, keep, slotA, pmaxA, out);
}
